// VQVAE_1348619730949
// MI455X (gfx1250) — matrix operand order repaired and checked by reading, not compiled, not run on hardware
//
#include <hip/hip_runtime.h>
#include <hip/hip_bf16.h>
#include <cstdint>

typedef _Float16 f16;
typedef __attribute__((ext_vector_type(16))) _Float16 v16h;
typedef __attribute__((ext_vector_type(8)))  _Float16 v8h;
typedef __attribute__((ext_vector_type(8)))  float    v8f;

union V16 { v16h v; v8h h[2]; };

__device__ __forceinline__ v8f wmma_f16(const V16& A, const V16& B, v8f C) {
    return __builtin_amdgcn_wmma_f32_16x16x32_f16(
        false, A.v, false, B.v, (short)0, C, false, false);
}

// Async global->LDS 16-byte copy (ASYNCcnt-tracked); per-lane LDS dest addr.
__device__ __forceinline__ void async_copy_b128(const f16* gsrc, f16* ldst) {
    unsigned lds = (unsigned)(uintptr_t)ldst;          // low 32 bits = LDS offset
    unsigned long long ga = (unsigned long long)(uintptr_t)gsrc;
    asm volatile("global_load_async_to_lds_b128 %0, %1, off"
                 :: "v"(lds), "v"(ga) : "memory");
}

__device__ __forceinline__ void wait_async0() {
#if __has_builtin(__builtin_amdgcn_s_wait_asynccnt)
    __builtin_amdgcn_s_wait_asynccnt(0);
#else
    asm volatile("s_wait_asynccnt 0" ::: "memory");
#endif
}

// ---------------------------------------------------------------------------
// Weight re-layout kernels (fp32 -> f16, [cout][K] with K = tap*CIN + cin)
// ---------------------------------------------------------------------------
__global__ void prep_w2t(const float* __restrict__ w, f16* __restrict__ o) {
    int e = blockIdx.x * 256 + threadIdx.x;           // 256*4096 = 1048576
    int cout = e >> 12, k = e & 4095;
    int tap = k >> 8, cin = k & 255;
    int ky = tap >> 2, kx = tap & 3;
    o[e] = (f16)w[((cout * 256 + cin) * 4 + ky) * 4 + kx];
}

__global__ void prep_w3t(const float* __restrict__ w, f16* __restrict__ o) {
    int e = blockIdx.x * 256 + threadIdx.x;           // 64*256 = 16384
    o[e] = (f16)w[e];                                  // [64][256][1][1] already [cout][cin]
}

__global__ void prep_wd1t(const float* __restrict__ w, f16* __restrict__ o) {
    int e = blockIdx.x * 256 + threadIdx.x;           // 256*576 = 147456
    int cout = e / 576, k = e % 576;
    int tap = k >> 6, cin = k & 63;
    int ky = tap / 3, kx = tap % 3;
    o[e] = (f16)w[((cout * 64 + cin) * 3 + ky) * 3 + kx];
}

// ConvT weights, per parity class pc = (Y&1)*2 + (X&1); tap j = jy*2+jx
__global__ void prep_wd2t(const float* __restrict__ w, f16* __restrict__ o) {
    int e = blockIdx.x * 256 + threadIdx.x;           // 4*256*1024 = 1048576
    int pc = e >> 18, r = e & 262143;
    int oc = r >> 10, k = r & 1023;
    int tap = k >> 8, ic = k & 255;
    int jy = tap >> 1, jx = tap & 1;
    int py = pc >> 1, px = pc & 1;
    int ky = py ? (jy ? 2 : 0) : (jy ? 3 : 1);
    int kx = px ? (jx ? 2 : 0) : (jx ? 3 : 1);
    o[e] = (f16)w[((ic * 256 + oc) * 4 + ky) * 4 + kx];
}

// d3 ConvT weights fp32, layout [pc][tap][i=256][o=3]
__global__ void prep_wd3t(const float* __restrict__ w, float* __restrict__ o) {
    int e = blockIdx.x * 256 + threadIdx.x;           // 4*4*256*3 = 12288
    int pc = e / 3072, r = e % 3072;
    int tap = r / 768, r2 = r % 768;
    int ic = r2 / 3, oc = r2 % 3;
    int jy = tap >> 1, jx = tap & 1;
    int py = pc >> 1, px = pc & 1;
    int ky = py ? (jy ? 2 : 0) : (jy ? 3 : 1);
    int kx = px ? (jx ? 2 : 0) : (jx ? 3 : 1);
    o[e] = w[((ic * 3 + oc) * 4 + ky) * 4 + kx];
}

// ---------------------------------------------------------------------------
// e1: 3->256, k4 s2 p1, NCHW fp32 in -> NHWC f16 out. One pixel per block.
// ---------------------------------------------------------------------------
__global__ __launch_bounds__(256) void conv_e1(
    const float* __restrict__ x, const float* __restrict__ w,
    const float* __restrict__ b, f16* __restrict__ out) {
    __shared__ float patch[48];
    int p = blockIdx.x;                                // 16*128*128 pixels
    int n = p >> 14, rem = p & 16383;
    int oy = rem >> 7, ox = rem & 127;
    int t = threadIdx.x;
    if (t < 48) {
        int c = t >> 4, r = t & 15, ky = r >> 2, kx = r & 3;
        int iy = 2 * oy - 1 + ky, ix = 2 * ox - 1 + kx;
        float v = 0.f;
        if (iy >= 0 && iy < 256 && ix >= 0 && ix < 256)
            v = x[(((size_t)n * 3 + c) * 256 + iy) * 256 + ix];
        patch[t] = v;
    }
    __syncthreads();
    float acc = b[t];
    const float* wp = w + t * 48;                      // OIHW contiguous per cout
    #pragma unroll
    for (int j = 0; j < 48; ++j) acc += patch[j] * wp[j];
    acc = fmaxf(acc, 0.f);
    out[(size_t)p * 256 + t] = (f16)acc;
}

// ---------------------------------------------------------------------------
// Generic implicit-GEMM WMMA conv. NHWC f16 in, weights [COUT][KH*KW*CIN] f16.
// 256 thr (8 waves) arranged WGM x WGN; each wave owns a 32x32 tile (2x2 WMMA,
// operands reused twice). Block tile = (WGM*32) couts x (WGN*32) pixels.
// K stepped by 32, double-buffered LDS staged via async global->LDS copies
// (ASYNCcnt) with one barrier per step.
// ---------------------------------------------------------------------------
template <int CIN, int COUT, int KH, int KW, int STR, int PAD,
          int HIN, int WIN, int HOUT, int WOUT, int WGM, int WGN,
          bool RELU, bool OUTF32>
__global__ __launch_bounds__(256) void conv_wmma(
    const f16* __restrict__ in, const f16* __restrict__ wT,
    const float* __restrict__ bias, f16* __restrict__ outH,
    float* __restrict__ outF) {
    constexpr int BM = WGM * 32, BN = WGN * 32;
    constexpr int KTOT = KH * KW * CIN;
    constexpr int KSTEPS = KTOT / 32;
    constexpr int KPT = CIN / 32;                      // K-steps per tap
    constexpr int SA = 40, SB = 40;                    // padded LDS row strides
    __shared__ __attribute__((aligned(16))) f16 Alds[2][BM * SA];
    __shared__ __attribute__((aligned(16))) f16 Blds[2][BN * SB];

    const int t = threadIdx.x;
    const int pixBase = blockIdx.x * BN;
    const int coutBase = blockIdx.y * BM;

    // --- A staging role: BM rows x 32 K per step ---
    constexpr int APT = BM / 8;                        // f16 per thread (8|16)
    constexpr int ATHR = 32 / APT;                     // threads per row
    const int aRow = t / ATHR, aSeg = (t % ATHR) * APT;
    const f16* aSrc = wT + (size_t)(coutBase + aRow) * KTOT + aSeg;

    // --- B staging role: BN pixels x 32 K per step (im2col gather) ---
    constexpr int BPT = BN / 8;                        // f16 per thread (8|16)
    constexpr int BTHR = 32 / BPT;                     // threads per pixel
    const int bPix = t / BTHR, bSeg = (t % BTHR) * BPT;
    int p = pixBase + bPix;
    int n = p / (HOUT * WOUT);
    int rp = p % (HOUT * WOUT);
    int oy = rp / WOUT, ox = rp % WOUT;

    // --- compute role ---
    const int lane = t & 31, wave = t >> 5;
    const int waveM = wave % WGM, waveN = wave / WGM;
    const int hi = lane >> 4, lr = lane & 15;
    const int aBase = (waveM * 32 + lr) * SA + hi * 8;
    const int bBase = (waveN * 32 + lr) * SB + hi * 8;

    v8f acc[2][2] = {};
    for (int kt = 0; kt < KSTEPS; ++kt) {
        const int pb = kt & 1;
        f16* Ab = Alds[pb];
        f16* Bb = Blds[pb];
        // stage A (weights; always in-bounds; L2-resident stream) via async DMA
        #pragma unroll
        for (int j = 0; j < APT / 8; ++j)
            async_copy_b128(aSrc + kt * 32 + j * 8, &Ab[aRow * SA + aSeg + j * 8]);
        __builtin_prefetch(aSrc + (kt + 4) * 32, 0, 1);   // global_prefetch_b8
        // stage B: async for in-bounds lanes, LDS zero-store for padding
        int tap = kt / KPT;
        int c0 = (kt % KPT) * 32;
        int ky = tap / KW, kx = tap % KW;
        int iy = oy * STR - PAD + ky, ix = ox * STR - PAD + kx;
        bool inb = (iy >= 0 && iy < HIN && ix >= 0 && ix < WIN);
        const f16* bp = in + (((size_t)n * HIN + iy) * WIN + ix) * CIN + c0 + bSeg;
        #pragma unroll
        for (int j = 0; j < BPT / 8; ++j) {
            if (inb) {
                async_copy_b128(bp + j * 8, &Bb[bPix * SB + bSeg + j * 8]);
            } else {
                v8h zv = {};
                *reinterpret_cast<v8h*>(&Bb[bPix * SB + bSeg + j * 8]) = zv;
            }
        }
        wait_async0();
        __syncthreads();
        V16 A0, A1, B0, B1;
        A0.h[0] = *reinterpret_cast<const v8h*>(&Ab[aBase]);
        A0.h[1] = *reinterpret_cast<const v8h*>(&Ab[aBase + 16]);
        A1.h[0] = *reinterpret_cast<const v8h*>(&Ab[aBase + 16 * SA]);
        A1.h[1] = *reinterpret_cast<const v8h*>(&Ab[aBase + 16 * SA + 16]);
        B0.h[0] = *reinterpret_cast<const v8h*>(&Bb[bBase]);
        B0.h[1] = *reinterpret_cast<const v8h*>(&Bb[bBase + 16]);
        B1.h[0] = *reinterpret_cast<const v8h*>(&Bb[bBase + 16 * SB]);
        B1.h[1] = *reinterpret_cast<const v8h*>(&Bb[bBase + 16 * SB + 16]);
        acc[0][0] = wmma_f16(A0, B0, acc[0][0]);
        acc[0][1] = wmma_f16(A0, B1, acc[0][1]);
        acc[1][0] = wmma_f16(A1, B0, acc[1][0]);
        acc[1][1] = wmma_f16(A1, B1, acc[1][1]);
    }

    // --- epilogue: lanes of one tile cover 8 consecutive couts -> b128 pack ---
    #pragma unroll
    for (int am = 0; am < 2; ++am) {
        #pragma unroll
        for (int bn = 0; bn < 2; ++bn) {
            const size_t pix = (size_t)pixBase + waveN * 32 + bn * 16 + lr;
            const int coutB = coutBase + waveM * 32 + am * 16 + hi * 8;
            if (OUTF32) {
                float4 lo, hic;
                float* dst = outF + pix * COUT + coutB;
                lo.x = acc[am][bn][0] + bias[coutB + 0];
                lo.y = acc[am][bn][1] + bias[coutB + 1];
                lo.z = acc[am][bn][2] + bias[coutB + 2];
                lo.w = acc[am][bn][3] + bias[coutB + 3];
                hic.x = acc[am][bn][4] + bias[coutB + 4];
                hic.y = acc[am][bn][5] + bias[coutB + 5];
                hic.z = acc[am][bn][6] + bias[coutB + 6];
                hic.w = acc[am][bn][7] + bias[coutB + 7];
                *reinterpret_cast<float4*>(dst) = lo;
                *reinterpret_cast<float4*>(dst + 4) = hic;
            } else {
                v8h ov;
                #pragma unroll
                for (int v = 0; v < 8; ++v) {
                    float val = acc[am][bn][v] + bias[coutB + v];
                    if (RELU) val = fmaxf(val, 0.f);
                    ov[v] = (f16)val;
                }
                *reinterpret_cast<v8h*>(outH + pix * COUT + coutB) = ov;
            }
        }
    }
}

// ---------------------------------------------------------------------------
// d2: ConvTranspose2d 256->256 k4 s2 p1 via 4 parity classes, WMMA. K=1024.
// in [16,64,64,256] f16, wT [4][256][1024] f16, out [16,128,128,256] f16 +ReLU
// Same 2x2-wave-tile / double-buffered async-staged scheme, block tile 128x64.
// ---------------------------------------------------------------------------
__global__ __launch_bounds__(256) void convt_wmma_d2(
    const f16* __restrict__ in, const f16* __restrict__ wT,
    const float* __restrict__ bias, f16* __restrict__ out) {
    constexpr int SA = 40, SB = 40;
    __shared__ __attribute__((aligned(16))) f16 Alds[2][128 * SA];
    __shared__ __attribute__((aligned(16))) f16 Blds[2][64 * SB];

    const int t = threadIdx.x;
    const int pixBase = blockIdx.x * 64;               // half-res pixels per class
    const int coutBase = blockIdx.y * 128;
    const int pc = blockIdx.z;
    const int py = pc >> 1, px = pc & 1;

    const int aRow = t >> 1, aSeg = (t & 1) * 16;      // 128 rows x 2 thr x 16 f16
    const f16* aSrc = wT + ((size_t)pc * 256 + coutBase + aRow) * 1024 + aSeg;

    const int bPix = t >> 2, bSeg = (t & 3) * 8;       // 64 pix x 4 thr x 8 f16
    int p = pixBase + bPix;
    int n = p >> 12, rp = p & 4095;
    int Yh = rp >> 6, Xh = rp & 63;

    const int lane = t & 31, wave = t >> 5;
    const int waveM = wave & 3, waveN = wave >> 2;
    const int hi = lane >> 4, lr = lane & 15;
    const int aBase = (waveM * 32 + lr) * SA + hi * 8;
    const int bBase = (waveN * 32 + lr) * SB + hi * 8;

    v8f acc[2][2] = {};
    for (int kt = 0; kt < 32; ++kt) {
        const int pb = kt & 1;
        f16* Ab = Alds[pb];
        f16* Bb = Blds[pb];
        async_copy_b128(aSrc + kt * 32,     &Ab[aRow * SA + aSeg]);
        async_copy_b128(aSrc + kt * 32 + 8, &Ab[aRow * SA + aSeg + 8]);
        __builtin_prefetch(aSrc + (kt + 4) * 32, 0, 1);
        int tap = kt >> 3, c0 = (kt & 7) * 32;
        int jy = tap >> 1, jx = tap & 1;
        int yi = Yh + (py ? (jy ? 0 : 1) : (jy ? -1 : 0));
        int xi = Xh + (px ? (jx ? 0 : 1) : (jx ? -1 : 0));
        if (yi >= 0 && yi < 64 && xi >= 0 && xi < 64) {
            async_copy_b128(
                in + (((size_t)n * 64 + yi) * 64 + xi) * 256 + c0 + bSeg,
                &Bb[bPix * SB + bSeg]);
        } else {
            v8h zv = {};
            *reinterpret_cast<v8h*>(&Bb[bPix * SB + bSeg]) = zv;
        }
        wait_async0();
        __syncthreads();
        V16 A0, A1, B0, B1;
        A0.h[0] = *reinterpret_cast<const v8h*>(&Ab[aBase]);
        A0.h[1] = *reinterpret_cast<const v8h*>(&Ab[aBase + 16]);
        A1.h[0] = *reinterpret_cast<const v8h*>(&Ab[aBase + 16 * SA]);
        A1.h[1] = *reinterpret_cast<const v8h*>(&Ab[aBase + 16 * SA + 16]);
        B0.h[0] = *reinterpret_cast<const v8h*>(&Bb[bBase]);
        B0.h[1] = *reinterpret_cast<const v8h*>(&Bb[bBase + 16]);
        B1.h[0] = *reinterpret_cast<const v8h*>(&Bb[bBase + 16 * SB]);
        B1.h[1] = *reinterpret_cast<const v8h*>(&Bb[bBase + 16 * SB + 16]);
        acc[0][0] = wmma_f16(A0, B0, acc[0][0]);
        acc[0][1] = wmma_f16(A0, B1, acc[0][1]);
        acc[1][0] = wmma_f16(A1, B0, acc[1][0]);
        acc[1][1] = wmma_f16(A1, B1, acc[1][1]);
    }

    #pragma unroll
    for (int am = 0; am < 2; ++am) {
        #pragma unroll
        for (int bn = 0; bn < 2; ++bn) {
            const int pp = pixBase + waveN * 32 + bn * 16 + lr;
            int n2 = pp >> 12, rp2 = pp & 4095;
            int Y = ((rp2 >> 6) << 1) + py, X = ((rp2 & 63) << 1) + px;
            const int coutB = coutBase + waveM * 32 + am * 16 + hi * 8;
            v8h ov;
            #pragma unroll
            for (int v = 0; v < 8; ++v) {
                float val = fmaxf(acc[am][bn][v] + bias[coutB + v], 0.f);
                ov[v] = (f16)val;
            }
            *reinterpret_cast<v8h*>(
                out + (((size_t)n2 * 128 + Y) * 128 + X) * 256 + coutB) = ov;
        }
    }
}

// ---------------------------------------------------------------------------
// VQ: per-pixel argmin over 512 codes (fp32), emit z_q f16 + deterministic
// per-block partial of sum((z_q - z_e)^2).
// ---------------------------------------------------------------------------
__global__ __launch_bounds__(256) void vq_kernel(
    const float* __restrict__ ze, const float* __restrict__ cb,
    f16* __restrict__ zq16, float* __restrict__ partials) {
    __shared__ __attribute__((aligned(16))) float cbl[128 * 64];
    __shared__ float red[256];
    const int t = threadIdx.x;
    const size_t pixel = (size_t)blockIdx.x * 256 + t;
    float z[64];
    const float* zp = ze + pixel * 64;
    #pragma unroll
    for (int k = 0; k < 64; ++k) z[k] = zp[k];

    float best = 3.4e38f;
    int bi = 0;
    for (int cc = 0; cc < 4; ++cc) {
        __syncthreads();
        const float4* src = reinterpret_cast<const float4*>(cb + cc * 128 * 64);
        float4* dst = reinterpret_cast<float4*>(cbl);
        #pragma unroll
        for (int j = 0; j < 8; ++j) dst[t + 256 * j] = src[t + 256 * j];
        __syncthreads();
        for (int c = 0; c < 128; ++c) {
            const float* cr = cbl + c * 64;
            float d = 0.f;
            #pragma unroll
            for (int k = 0; k < 64; ++k) { float dv = z[k] - cr[k]; d += dv * dv; }
            if (d < best) { best = d; bi = cc * 128 + c; }
        }
    }
    const float* cq = cb + (size_t)bi * 64;
    f16* qo = zq16 + pixel * 64;
    float sq = 0.f;
    #pragma unroll
    for (int k = 0; k < 64; ++k) {
        float q = cq[k];
        float dv = q - z[k];
        sq += dv * dv;
        qo[k] = (f16)q;
    }
    red[t] = sq;
    __syncthreads();
    for (int s = 128; s > 0; s >>= 1) {
        if (t < s) red[t] += red[t + s];
        __syncthreads();
    }
    if (t == 0) partials[blockIdx.x] = red[0];
}

__global__ void loss_final(const float* __restrict__ partials,
                           float* __restrict__ lossOut) {
    __shared__ float red[256];
    int t = threadIdx.x;
    red[t] = partials[t];
    __syncthreads();
    for (int s = 128; s > 0; s >>= 1) {
        if (t < s) red[t] += red[t + s];
        __syncthreads();
    }
    if (t == 0) lossOut[0] = red[0] * 1.25f / (65536.f * 64.f);
}

// ---------------------------------------------------------------------------
// d3: ConvTranspose2d 256->3 k4 s2 p1 + sigmoid, VALU. out NCHW fp32.
// ---------------------------------------------------------------------------
__global__ __launch_bounds__(256) void convt_d3(
    const f16* __restrict__ in, const float* __restrict__ wT,
    const float* __restrict__ bias, float* __restrict__ out) {
    int tid = blockIdx.x * 256 + threadIdx.x;          // 16*256*256 = 1048576
    int n = tid >> 16, rem = tid & 65535;
    int Y = rem >> 8, X = rem & 255;
    int py = Y & 1, px = X & 1, pc = py * 2 + px;
    int Yh = Y >> 1, Xh = X >> 1;
    float a0 = bias[0], a1 = bias[1], a2 = bias[2];
    #pragma unroll
    for (int tap = 0; tap < 4; ++tap) {
        int jy = tap >> 1, jx = tap & 1;
        int yi = Yh + (py ? (jy ? 0 : 1) : (jy ? -1 : 0));
        int xi = Xh + (px ? (jx ? 0 : 1) : (jx ? -1 : 0));
        if (yi < 0 || yi >= 128 || xi < 0 || xi >= 128) continue;
        const f16* ip = in + (((size_t)n * 128 + yi) * 128 + xi) * 256;
        const float* wp = wT + (size_t)(pc * 4 + tap) * 256 * 3;
        for (int i = 0; i < 256; ++i) {
            float v = (float)ip[i];
            a0 += v * wp[i * 3 + 0];
            a1 += v * wp[i * 3 + 1];
            a2 += v * wp[i * 3 + 2];
        }
    }
    size_t ob = (size_t)n * 3 * 65536 + (size_t)Y * 256 + X;
    out[ob]          = 1.f / (1.f + expf(-a0));
    out[ob + 65536]  = 1.f / (1.f + expf(-a1));
    out[ob + 131072] = 1.f / (1.f + expf(-a2));
}

// ---------------------------------------------------------------------------
// Launch
// ---------------------------------------------------------------------------
extern "C" void kernel_launch(void* const* d_in, const int* in_sizes, int n_in,
                              void* d_out, int out_size, void* d_ws, size_t ws_size,
                              hipStream_t stream) {
    const float* x        = (const float*)d_in[0];
    const float* ew1      = (const float*)d_in[1];
    const float* eb1      = (const float*)d_in[2];
    const float* ew2      = (const float*)d_in[3];
    const float* eb2      = (const float*)d_in[4];
    const float* ew3      = (const float*)d_in[5];
    const float* eb3      = (const float*)d_in[6];
    const float* codebook = (const float*)d_in[7];
    const float* dw1      = (const float*)d_in[8];
    const float* db1      = (const float*)d_in[9];
    const float* dw2      = (const float*)d_in[10];
    const float* db2      = (const float*)d_in[11];
    const float* dw3      = (const float*)d_in[12];
    const float* db3      = (const float*)d_in[13];
    float* out = (float*)d_out;

    char* ws = (char*)d_ws;
    f16*   h1    = (f16*)  (ws);                       // 16*128*128*256 f16
    f16*   w2t   = (f16*)  (ws + 134217728ull);        // 256*4096 f16
    f16*   h2    = (f16*)  (ws + 136314880ull);        // 16*64*64*256 f16
    f16*   w3t   = (f16*)  (ws + 169869312ull);        // 64*256 f16
    float* z_e   = (float*)(ws + 169902080ull);        // 65536*64 f32
    f16*   zq16  = (f16*)  (ws + 186679296ull);        // 65536*64 f16
    f16*   wd1t  = (f16*)  (ws + 195067904ull);        // 256*576 f16
    f16*   d1o   = (f16*)  (ws + 195362816ull);        // 16*64*64*256 f16
    f16*   wd2t  = (f16*)  (ws + 228917248ull);        // 4*256*1024 f16
    f16*   d2o   = (f16*)  (ws + 231014400ull);        // 16*128*128*256 f16
    float* wd3t  = (float*)(ws + 365232128ull);        // 4*4*256*3 f32
    float* parts = (float*)(ws + 365281280ull);        // 256 f32

    // weight prep
    prep_w2t <<<4096, 256, 0, stream>>>(ew2, w2t);
    prep_w3t <<<64,   256, 0, stream>>>(ew3, w3t);
    prep_wd1t<<<576,  256, 0, stream>>>(dw1, wd1t);
    prep_wd2t<<<4096, 256, 0, stream>>>(dw2, wd2t);
    prep_wd3t<<<48,   256, 0, stream>>>(dw3, wd3t);

    // encoder
    conv_e1<<<262144, 256, 0, stream>>>(x, ew1, eb1, h1);
    conv_wmma<256, 256, 4, 4, 2, 1, 128, 128, 64, 64, 4, 2, true, false>
        <<<dim3(1024, 2), 256, 0, stream>>>(h1, w2t, eb2, h2, nullptr);
    conv_wmma<256, 64, 1, 1, 1, 0, 64, 64, 64, 64, 2, 4, false, true>
        <<<dim3(512, 1), 256, 0, stream>>>(h2, w3t, eb3, nullptr, z_e);

    // vector quantizer
    vq_kernel<<<256, 256, 0, stream>>>(z_e, codebook, zq16, parts);

    // decoder
    conv_wmma<64, 256, 3, 3, 1, 1, 64, 64, 64, 64, 4, 2, true, false>
        <<<dim3(1024, 2), 256, 0, stream>>>(zq16, wd1t, db1, d1o, nullptr);
    convt_wmma_d2<<<dim3(1024, 2, 4), 256, 0, stream>>>(d1o, wd2t, db2, d2o);
    convt_d3<<<4096, 256, 0, stream>>>(d2o, wd3t, db3, out);

    // scalar loss at end of output buffer
    loss_final<<<1, 256, 0, stream>>>(parts, out + (out_size - 1));
}
